// ZeroSAttention_9921374453883
// MI455X (gfx1250) — matrix operand order repaired and checked by reading, not compiled, not run on hardware
//
#include <hip/hip_runtime.h>
#include <hip/hip_bf16.h>
#include <math.h>

typedef __attribute__((ext_vector_type(16))) __bf16 v16bf;
typedef __attribute__((ext_vector_type(8)))  float  v8f;
typedef __attribute__((ext_vector_type(4)))  unsigned int u32x4;
typedef __attribute__((ext_vector_type(8)))  int    i32x8;
typedef __attribute__((ext_vector_type(4)))  int    i32x4;

#define L_SEQ  1024
#define DMODEL 768
#define NHEAD  12
#define DHEAD  64
#define GATE_LD 48     // padded row stride for the 36-wide gate projection

// LDS row stride in floats after TDM padding: 32 data dwords + 8 pad dwords
#define TILE_K   32
#define LDS_STR  40

// ---------------------------------------------------------------------------
// Tensor Data Mover: issue a 2-D tile load (Global fp32 -> LDS) per ISA D#
// layout (CDNA5 ISA). 8 pad dwords inserted every 32
// dwords (one tile row) -> LDS row stride = 40 dwords (bank spreading).
// Group0: count=1 | lds_addr | global_addr | type=2
// Group1: data_size=4B(2), pad_enable, pad_interval=4 (32dw), pad_amount=7 (8dw),
//         tensor_dim0/1, tile_dim0/1, tensor_dim0_stride
// Groups2/3: zero (2-D tensor). Rows past tensor_dim1 read as zero (OOB rule),
// which handles the 36-row gate weight with a 64-row tile for free.
// ---------------------------------------------------------------------------
__device__ __forceinline__ void tdm_load_tile_f32(const float* gsrc, void* lds_dst,
                                                  int tensor_d0, int tensor_d1,
                                                  int row_stride_elems,
                                                  int tile_d0, int tile_d1)
{
    unsigned long long ga = (unsigned long long)(uintptr_t)gsrc;
    unsigned lds = (unsigned)(uintptr_t)lds_dst;   // flat addr low 32 bits = LDS offset

    u32x4 g0;
    g0[0] = 1u;                                            // count=1 (valid user D#)
    g0[1] = lds;                                           // lds_addr (bytes)
    g0[2] = (unsigned)ga;                                  // global_addr[31:0]
    g0[3] = (unsigned)((ga >> 32) & 0x01ffffffu) | (2u << 30);  // ga[56:32] | type=2

    i32x8 g1;
    g1[0] = (int)((2u << 16) | (1u << 20) | (4u << 22) | (7u << 25));
    g1[1] = (int)(((unsigned)tensor_d0 & 0xffffu) << 16);  // tensor_dim0[15:0]
    g1[2] = (int)((((unsigned)tensor_d0) >> 16) |
                  (((unsigned)tensor_d1 & 0xffffu) << 16));// dim0[31:16] | dim1[15:0]
    g1[3] = (int)((((unsigned)tensor_d1) >> 16) |
                  (((unsigned)tile_d0) << 16));            // dim1[31:16] | tile_dim0
    g1[4] = (int)((unsigned)tile_d1);                      // tile_dim1 (tile_dim2=0)
    g1[5] = (int)(unsigned)row_stride_elems;               // tensor_dim0_stride[31:0]
    g1[6] = 0;
    g1[7] = 0;

    i32x4 gz; gz[0] = 0; gz[1] = 0; gz[2] = 0; gz[3] = 0;
#if defined(__clang_major__) && (__clang_major__ >= 23)
    i32x8 gz8; gz8[0]=0; gz8[1]=0; gz8[2]=0; gz8[3]=0; gz8[4]=0; gz8[5]=0; gz8[6]=0; gz8[7]=0;
    __builtin_amdgcn_tensor_load_to_lds(g0, g1, gz, gz, gz8, 0);
#else
    __builtin_amdgcn_tensor_load_to_lds(g0, g1, gz, gz, 0);
#endif
}

// ---------------------------------------------------------------------------
// C[M x N] = A[M x K] @ W[N x K]^T, fp32 in/out, bf16 WMMA core.
// Block = 256 threads = 8 waves. Block tile: 128 (M) x 64 (N).
// Wave w owns M-strip [w*16, w*16+16); runs 4 WMMAs/K-step (A frag reused).
// K-slabs (32 wide) staged in LDS by the Tensor Data Mover, double buffered:
//   iter i: wave0 s_wait_tensorcnt(0) -> barrier -> wave0 issues slab i+1 DMA
//           -> all waves compute slab i from LDS.
// Fragment layouts per CDNA5 ISA 7.12.2 (wave32):
//   A (16x32): lane<16 -> M=lane, K={0..7},{16..23}; lane>=16 -> M=lane-16,
//              K={8..15},{24..31}   (elem j<8 -> K=hi*8+j; j>=8 -> 16+hi*8+j-8)
//   B (32x16): lane -> N=lane&15, elem j -> K as for A
//   C (16x16): elem r -> M=(lane<16 ? r : 8+r), N=lane&15
// ---------------------------------------------------------------------------
__global__ __launch_bounds__(256)
void gemm_xwT_wmma(const float* __restrict__ A, const float* __restrict__ W,
                   float* __restrict__ C, int M, int Nvalid, int Kd, int ldc)
{
    __shared__ float Abuf[2][128 * LDS_STR];   // 2 x 20 KB
    __shared__ float Bbuf[2][64 * LDS_STR];    // 2 x 10 KB

    const int tid  = threadIdx.x;
    const int lane = tid & 31;
    const int wave = tid >> 5;                 // 0..7 -> M sub-strip
    const int m0   = blockIdx.x * 128;
    const int n0   = blockIdx.y * 64;

    const int l    = lane & 15;
    const int hi   = lane >> 4;
    const int arow = wave * 16 + l;            // row of A tile this lane feeds

    v8f acc[4];
#pragma unroll
    for (int nt = 0; nt < 4; ++nt) acc[nt] = (v8f){};

    const int nk = Kd / TILE_K;

    // prologue: stage slab 0 into buffer 0 (wave 0 only; TDM ignores EXEC,
    // so issue from exactly one wave)
    if (wave == 0) {
        tdm_load_tile_f32(A + (size_t)m0 * Kd, &Abuf[0][0],
                          Kd, M - m0, Kd, TILE_K, 128);
        tdm_load_tile_f32(W + (size_t)n0 * Kd, &Bbuf[0][0],
                          Kd, Nvalid - n0, Kd, TILE_K, 64);
    }

    for (int i = 0; i < nk; ++i) {
        const int b = i & 1;
        if (wave == 0) __builtin_amdgcn_s_wait_tensorcnt(0);
        __syncthreads();                        // slab i visible to all waves;
                                                // also: everyone done reading buf b^1
        if (wave == 0 && (i + 1) < nk) {
            const int k1 = (i + 1) * TILE_K;
            tdm_load_tile_f32(A + (size_t)m0 * Kd + k1, &Abuf[b ^ 1][0],
                              Kd - k1, M - m0, Kd, TILE_K, 128);
            tdm_load_tile_f32(W + (size_t)n0 * Kd + k1, &Bbuf[b ^ 1][0],
                              Kd - k1, Nvalid - n0, Kd, TILE_K, 64);
        }

        const float* Ab = &Abuf[b][0];
        const float* Bb = &Bbuf[b][0];

        v16bf af;
        const int ka = hi * 8;
#pragma unroll
        for (int j = 0; j < 8; ++j) {
            af[j]     = (__bf16)Ab[arow * LDS_STR + ka + j];
            af[j + 8] = (__bf16)Ab[arow * LDS_STR + ka + 16 + j];
        }
#pragma unroll
        for (int nt = 0; nt < 4; ++nt) {
            v16bf bfm;
#pragma unroll
            for (int j = 0; j < 8; ++j) {
                bfm[j]     = (__bf16)Bb[(nt * 16 + l) * LDS_STR + ka + j];
                bfm[j + 8] = (__bf16)Bb[(nt * 16 + l) * LDS_STR + ka + 16 + j];
            }
            acc[nt] = __builtin_amdgcn_wmma_f32_16x16x32_bf16(false, af, false, bfm,
                                                              (short)0, acc[nt],
                                                              false, false);
        }
        // no trailing barrier: next iteration's post-wait barrier orders reuse
    }

    const int col   = lane & 15;
    const int rbase = hi * 8;
#pragma unroll
    for (int nt = 0; nt < 4; ++nt) {
#pragma unroll
        for (int r = 0; r < 8; ++r) {
            const int gm = m0 + wave * 16 + rbase + r;
            const int gn = n0 + nt * 16 + col;
            if (gm < M && gn < Nvalid) C[(size_t)gm * ldc + gn] = acc[nt][r];
        }
    }
}

// ---------------------------------------------------------------------------
// s_i pre-pass: one block (64 threads = 2 waves) per head.
// ---------------------------------------------------------------------------
__global__ __launch_bounds__(64)
void compute_s_kernel(const float* __restrict__ u, const float* __restrict__ prior_mu,
                      const float* __restrict__ prior_log_tau,
                      float* __restrict__ s_i, float* __restrict__ s_max)
{
    const int head = blockIdx.x;
    const int dd   = threadIdx.x;
    __shared__ float xw[2];

    float pl = prior_log_tau[head];
    pl = fminf(fmaxf(pl, -50.f), 30.f);
    const float tau = expf(pl);
    const float mu  = prior_mu[head * DHEAD + dd];

    float usum = 0.f;
    float smax = -3.402823466e38f;

    for (int t = 0; t < L_SEQ; ++t) {
        const float uv = u[(size_t)t * DMODEL + head * DHEAD + dd];
        usum += uv;
        const float bar = (tau * mu + usum) / (tau + (float)(t + 1));
        float p = uv * bar;
        for (int m = 16; m >= 1; m >>= 1) p += __shfl_xor(p, m, 32);
        if ((threadIdx.x & 31) == 0) xw[threadIdx.x >> 5] = p;
        __syncthreads();
        if (threadIdx.x == 0) {
            const float s = -(xw[0] + xw[1]) * 0.125f;
            s_i[(size_t)t * NHEAD + head] = s;
            smax = fmaxf(smax, s);
        }
        __syncthreads();
    }
    if (threadIdx.x == 0) s_max[head] = smax;
}

// ---------------------------------------------------------------------------
// RoPE + L2 normalization of q and k, in place. One block per (t, head).
// ---------------------------------------------------------------------------
__global__ __launch_bounds__(64)
void rope_norm_kernel(float* __restrict__ q, float* __restrict__ k)
{
    const int t    = blockIdx.x / NHEAD;
    const int head = blockIdx.x % NHEAD;
    const int j    = threadIdx.x;
    __shared__ float xw[4];

    const size_t base = (size_t)t * DMODEL + head * DHEAD;
    const float inv = __powf(10000.f, -(float)(j & ~1) / (float)DHEAD);
    const float ang = (float)t * inv;
    const float c = __cosf(ang), s = __sinf(ang);

    const float qv = q[base + j];
    const float kv = k[base + j];
    const float qp = (j < 32) ? -q[base + j + 32] : q[base + j - 32];
    const float kp = (j < 32) ? -k[base + j + 32] : k[base + j - 32];
    const float qr = qv * c + qp * s;
    const float kr = kv * c + kp * s;

    float nq = qr * qr, nk = kr * kr;
    for (int m = 16; m >= 1; m >>= 1) {
        nq += __shfl_xor(nq, m, 32);
        nk += __shfl_xor(nk, m, 32);
    }
    if ((j & 31) == 0) { xw[j >> 5] = nq; xw[2 + (j >> 5)] = nk; }
    __syncthreads();
    nq = sqrtf(xw[0] + xw[1]);
    nk = sqrtf(xw[2] + xw[3]);
    q[base + j] = qr / fmaxf(nq, 1e-12f);
    k[base + j] = kr / fmaxf(nk, 1e-12f);
}

// ---------------------------------------------------------------------------
// Main causal scan: one block (256 threads) per head, sequential over L.
// F,G,H (64x64 each) register-resident; fused head-dim LayerNorm.
// ---------------------------------------------------------------------------
__global__ __launch_bounds__(256)
void scan_kernel(const float* __restrict__ qn, const float* __restrict__ kn,
                 const float* __restrict__ v,  const float* __restrict__ s_i,
                 const float* __restrict__ s_max, const float* __restrict__ gate,
                 float* __restrict__ attn)
{
    const int head = blockIdx.x;
    const int tid  = threadIdx.x;
    const int e    = tid & 63;
    const int dg   = tid >> 6;

    __shared__ float ks[64], vs[64], qs[64];
    __shared__ float part[4 * 64];
    __shared__ float outs[64], r1[64], r2[64];
    __shared__ float sc[8];

    float F[16], G[16], H[16];
#pragma unroll
    for (int j = 0; j < 16; ++j) F[j] = G[j] = H[j] = 0.f;

    float E = 0.f, P = 0.f;
    const float smax = s_max[head];
    const float eps  = 1e-12f;

    for (int t = 0; t < L_SEQ; ++t) {
        __syncthreads();
        if (tid < 64) {
            const size_t b = (size_t)t * DMODEL + head * DHEAD + tid;
            ks[tid] = kn[b]; vs[tid] = v[b]; qs[tid] = qn[b];
        }
        if (tid == 0) {
            const float s  = s_i[(size_t)t * NHEAD + head];
            const float es = expf(s - smax);
            E += es; P += s;
            const float g0 = gate[(size_t)t * GATE_LD + head * 3 + 0];
            const float g1 = gate[(size_t)t * GATE_LD + head * 3 + 1];
            const float sig1 = 1.f / (1.f + expf(-g0));
            const float sigh = 1.f / (1.f + expf(-g1));
            const float tt = (float)(t + 1);
            const float alpha = sigh / (E + eps);
            const float beta  = (sig1 - sigh) / tt;
            const float gamma = -((beta * P + sigh) / tt);
            sc[0] = alpha; sc[1] = beta; sc[2] = gamma; sc[3] = es; sc[4] = s;
        }
        __syncthreads();

        const float alpha = sc[0], beta = sc[1], gamma = sc[2];
        const float es = sc[3], s = sc[4];
        const float ve = vs[e];
        float acc = 0.f;
#pragma unroll
        for (int j = 0; j < 16; ++j) {
            const int dd = dg * 16 + j;
            const float kv = ks[dd] * ve;
            F[j] += es * kv;
            G[j] += s  * kv;
            H[j] += kv;
            acc += qs[dd] * (alpha * F[j] + beta * G[j] + gamma * H[j]);
        }
        part[dg * 64 + e] = acc;
        __syncthreads();

        if (tid < 64) {
            const float o = part[e] + part[64 + e] + part[128 + e] + part[192 + e];
            outs[e] = o; r1[e] = o; r2[e] = o * o;
        }
        __syncthreads();
        for (int str = 32; str >= 1; str >>= 1) {
            if (tid < str) { r1[tid] += r1[tid + str]; r2[tid] += r2[tid + str]; }
            __syncthreads();
        }
        if (tid < 64) {
            const float mu  = r1[0] * (1.f / 64.f);
            const float var = r2[0] * (1.f / 64.f) - mu * mu;
            attn[(size_t)t * DMODEL + head * DHEAD + e] =
                (outs[e] - mu) * rsqrtf(var + 1e-5f);
        }
    }
}

// ---------------------------------------------------------------------------
extern "C" void kernel_launch(void* const* d_in, const int* in_sizes, int n_in,
                              void* d_out, int out_size, void* d_ws, size_t ws_size,
                              hipStream_t stream)
{
    const float* X    = (const float*)d_in[0];
    const float* Wq   = (const float*)d_in[1];
    const float* Wk   = (const float*)d_in[2];
    const float* Wv   = (const float*)d_in[3];
    const float* Wu   = (const float*)d_in[4];
    const float* Wg   = (const float*)d_in[5];
    const float* Wo   = (const float*)d_in[6];
    const float* pmu  = (const float*)d_in[7];
    const float* ptau = (const float*)d_in[8];

    float* ws = (float*)d_ws;
    const size_t LD = (size_t)L_SEQ * DMODEL;
    float* q    = ws;
    float* k    = q + LD;
    float* v    = k + LD;
    float* u    = v + LD;
    float* gate = u + LD;                              // L*48 (36 valid)
    float* s_i  = gate + (size_t)L_SEQ * GATE_LD;
    float* smax = s_i + (size_t)L_SEQ * NHEAD;
    float* attn = smax + 32;

    dim3 blk(256);
    dim3 gproj(L_SEQ / 128, DMODEL / 64);              // 8 x 12

    gemm_xwT_wmma<<<gproj, blk, 0, stream>>>(X, Wq, q, L_SEQ, DMODEL, DMODEL, DMODEL);
    gemm_xwT_wmma<<<gproj, blk, 0, stream>>>(X, Wk, k, L_SEQ, DMODEL, DMODEL, DMODEL);
    gemm_xwT_wmma<<<gproj, blk, 0, stream>>>(X, Wv, v, L_SEQ, DMODEL, DMODEL, DMODEL);
    gemm_xwT_wmma<<<gproj, blk, 0, stream>>>(X, Wu, u, L_SEQ, DMODEL, DMODEL, DMODEL);

    dim3 ggate(L_SEQ / 128, 1);
    gemm_xwT_wmma<<<ggate, blk, 0, stream>>>(X, Wg, gate, L_SEQ, 3 * NHEAD, DMODEL, GATE_LD);

    compute_s_kernel<<<NHEAD, 64, 0, stream>>>(u, pmu, ptau, s_i, smax);
    rope_norm_kernel<<<L_SEQ * NHEAD, 64, 0, stream>>>(q, k);
    scan_kernel<<<NHEAD, 256, 0, stream>>>(q, k, v, s_i, smax, gate, attn);

    gemm_xwT_wmma<<<gproj, blk, 0, stream>>>(attn, Wo, (float*)d_out,
                                             L_SEQ, DMODEL, DMODEL, DMODEL);
}
